// MultiHeadAttention_83133386981783
// MI455X (gfx1250) — matrix operand order repaired and checked by reading, not compiled, not run on hardware
//
#include <hip/hip_runtime.h>

// Problem constants (fixed by the reference)
#define B_    4
#define N_    20000
#define K_    16      // neighbors
#define DIN   128
#define DH    64      // NHEAD*DV = 8*8
#define DOUT  64

typedef __attribute__((ext_vector_type(2)))  float   v2f;
typedef __attribute__((ext_vector_type(4)))  float   v4f;
typedef __attribute__((ext_vector_type(8)))  float   v8f;
typedef __attribute__((ext_vector_type(16))) __bf16  v16bf;
typedef __attribute__((ext_vector_type(8)))  __bf16  v8bf;

// D(16x16 f32) = A(16x32 bf16) * B(32x16 bf16) + C
#define WMMA_BF16(a, b, c) \
    __builtin_amdgcn_wmma_f32_16x16x32_bf16(false, (a), false, (b), (short)0, (c), false, false)

// Padded column stride for weight LDS (bf16 elems). 136*2B = 272B = 68 dwords
// -> bank step 4 per column: conflict-free half-wave b128 reads.
#define COLP 136

// LDS layout (bytes):
//   WkH | WkL | WvH | WvL : 4 * 64*COLP bf16 = 4*17408 = 69632 B
//   per-wave f32 staging  : 8 waves * 2304 floats = 73728 B
//   total 143360 B (dynamic; 2 workgroups per 320KB WGP)
#define SMEM_BYTES (4 * 64 * COLP * 2 + 8 * 2304 * 4)

__device__ __forceinline__ v16bf lds_bfrag(const __bf16* p) {
    v8bf lo = *(const v8bf*)(p);       // 16B aligned by construction
    v8bf hi = *(const v8bf*)(p + 8);
    return __builtin_shufflevector(lo, hi, 0,1,2,3,4,5,6,7,8,9,10,11,12,13,14,15);
}

__global__ __launch_bounds__(256)
void mha_wmma_kernel(const float* __restrict__ agent,
                     const float* __restrict__ neigh,
                     const float* __restrict__ Wq, const float* __restrict__ bq,
                     const float* __restrict__ Wk, const float* __restrict__ bk,
                     const float* __restrict__ Wv, const float* __restrict__ bv,
                     const float* __restrict__ Wo, const float* __restrict__ bo,
                     float* __restrict__ out,     // (B*N, 64)
                     float* __restrict__ att_out) // (B*N, 8, 16)
{
    extern __shared__ char smemraw[];
    __bf16* WkH = (__bf16*)smemraw;           // [n*COLP + k], column-major
    __bf16* WkL = WkH + 64 * COLP;
    __bf16* WvH = WkL + 64 * COLP;
    __bf16* WvL = WvH + 64 * COLP;
    float*  stgbase = (float*)(WvL + 64 * COLP);

    const int tid  = threadIdx.x;
    const int wid  = tid >> 5;
    const int lane = tid & 31;

    float* stg = stgbase + wid * 2304;
    float* Kst = stg;          // 16 x 64
    float* Vst = stg + 1024;   // 16 x 64
    float* Qst = stg + 2048;   // 64
    float* Ast = stg + 2112;   // 8 x 16
    float* Cst = stg + 2240;   // 64

    // ---- stage Wk, Wv into LDS as bf16 hi/lo split, column-major padded ----
    for (int e = tid; e < DIN * DH; e += 256) {
        int k = e >> 6, n = e & 63;
        float wk = Wk[e], wv = Wv[e];
        __bf16 kh = (__bf16)wk, vh = (__bf16)wv;
        WkH[n * COLP + k] = kh;
        WkL[n * COLP + k] = (__bf16)(wk - (float)kh);
        WvH[n * COLP + k] = vh;
        WvL[n * COLP + k] = (__bf16)(wv - (float)vh);
    }
    __syncthreads();

    const int m    = lane & 15;   // A-matrix row (neighbor index)
    const int half = lane >> 4;   // K sub-block select
    const int nl   = lane & 15;   // B/C column within 16-wide tile
    const int c0   = lane * 2;    // 2 output columns per lane for VALU parts

    for (int i = 0; i < 2; ++i) {
        const long p = (long)blockIdx.x * 16 + wid * 2 + i;   // flat node id, exact fit
        const float* Arow = neigh + p * (long)(K_ * DIN) + m * DIN;
        const float* ag   = agent + p * (long)DIN;

        // prefetch next node's 8KB neighbor tile while we crunch this one
        {
            long pn = p + 1; if (pn > (long)B_ * N_ - 1) pn = (long)B_ * N_ - 1;
            const float* nf = neigh + pn * (long)(K_ * DIN);
            __builtin_prefetch(nf + lane * 64, 0, 1);
            __builtin_prefetch(nf + lane * 64 + 32, 0, 1);
        }

        // ===== fused K/V projections (bf16x3 WMMA) + Q projection (VALU,
        //       co-executes with the XDL pipe) over the same K=32 chunks.
        //       Rolled loop: one chunk body = 24 WMMAs + 32 ds_load_b128 of
        //       independent work; keeps VGPRs < 256 (no s_set_vgpr_msb). =====
        v8f zero = {0,0,0,0,0,0,0,0};
        v8f ck[4], cv[4];
        #pragma unroll
        for (int nt = 0; nt < 4; ++nt) { ck[nt] = zero; cv[nt] = zero; }
        v2f q = *(const v2f*)(bq + c0);

        #pragma unroll 1
        for (int c = 0; c < 4; ++c) {               // 4 chunks of K=32
            const int k0 = c * 32;
            // --- A fragment: lane holds K = k0 + 16h + {0..15} ---
            const float* ax = Arow + k0 + 16 * half;
            v16bf ah, al;
            #pragma unroll
            for (int j = 0; j < 8; ++j) {
                float x = ax[j];
                float y = ax[8 + j];
                __bf16 xh = (__bf16)x, yh = (__bf16)y;
                ah[j]     = xh;  al[j]     = (__bf16)(x - (float)xh);
                ah[8 + j] = yh;  al[8 + j] = (__bf16)(y - (float)yh);
            }
            // --- B fragments: lane holds K = k0 + 16h + {0..15} of column n ---
            const int boff = k0 + 16 * half;
            #pragma unroll
            for (int nt = 0; nt < 4; ++nt) {
                const int col = nt * 16 + nl;
                v16bf bkh = lds_bfrag(WkH + col * COLP + boff);
                v16bf bkl = lds_bfrag(WkL + col * COLP + boff);
                ck[nt] = WMMA_BF16(ah, bkh, ck[nt]);
                ck[nt] = WMMA_BF16(ah, bkl, ck[nt]);
                ck[nt] = WMMA_BF16(al, bkh, ck[nt]);
                v16bf bvh = lds_bfrag(WvH + col * COLP + boff);
                v16bf bvl = lds_bfrag(WvL + col * COLP + boff);
                cv[nt] = WMMA_BF16(ah, bvh, cv[nt]);
                cv[nt] = WMMA_BF16(ah, bvl, cv[nt]);
                cv[nt] = WMMA_BF16(al, bvh, cv[nt]);
            }
            // --- Q partial for this K chunk: independent VALU work that the
            //     scheduler interleaves with the WMMA chain above ---
            #pragma unroll 8
            for (int kq = k0; kq < k0 + 32; ++kq) {
                float aa = ag[kq];                           // wave-uniform broadcast
                v2f  w  = *(const v2f*)(Wq + kq * DH + c0);  // coalesced
                q += aa * w;                                 // v_pk_fma_f32
            }
        }

        // bias add + write C tiles to LDS staging in [k][dh] layout
        {
            float bkb[4] = { bk[nl], bk[16 + nl], bk[32 + nl], bk[48 + nl] };
            float bvb[4] = { bv[nl], bv[16 + nl], bv[32 + nl], bv[48 + nl] };
            #pragma unroll
            for (int r = 0; r < 8; ++r) {
                int row = r + 8 * half;              // C layout: M = r + 8*(lane>>4)
                #pragma unroll
                for (int nt = 0; nt < 4; ++nt) {
                    Kst[row * 64 + nt * 16 + nl] = ck[nt][r] + bkb[nt];
                    Vst[row * 64 + nt * 16 + nl] = cv[nt][r] + bvb[nt];
                }
            }
            Qst[c0] = q.x; Qst[c0 + 1] = q.y;
        }
        __builtin_amdgcn_wave_barrier();

        // ===== scores + softmax: head h = lane>>2, 4 neighbors per lane =====
        {
            const int h  = lane >> 2;
            const int j0 = lane & 3;
            v4f q0 = *(const v4f*)(Qst + h * 8);       // preload Q into regs (2x b128)
            v4f q1 = *(const v4f*)(Qst + h * 8 + 4);
            float sc[4];
            #pragma unroll
            for (int j = 0; j < 4; ++j) {
                int kn = j0 + 4 * j;
                v4f k0v = *(const v4f*)(Kst + kn * 64 + h * 8);
                v4f k1v = *(const v4f*)(Kst + kn * 64 + h * 8 + 4);
                v4f d = q0 * k0v + q1 * k1v;
                sc[j] = (d.x + d.y + d.z + d.w) * 0.35355339059327379f;  // 1/sqrt(DV)
            }
            float mx = fmaxf(fmaxf(sc[0], sc[1]), fmaxf(sc[2], sc[3]));
            mx = fmaxf(mx, __shfl_xor(mx, 1, 4));
            mx = fmaxf(mx, __shfl_xor(mx, 2, 4));
            float ee[4]; float sum = 0.f;
            #pragma unroll
            for (int j = 0; j < 4; ++j) { ee[j] = expf(sc[j] - mx); sum += ee[j]; }
            sum += __shfl_xor(sum, 1, 4);
            sum += __shfl_xor(sum, 2, 4);
            float inv = 1.f / sum;
            float* ao = att_out + (p * 8 + h) * 16;
            #pragma unroll
            for (int j = 0; j < 4; ++j) {
                int kn = j0 + 4 * j;
                float a = ee[j] * inv;
                ao[kn] = a;                                   // second output: att
                Ast[h * 16 + kn] = a;
            }
        }
        __builtin_amdgcn_wave_barrier();

        // ===== ctx = att @ V  (2 cols per lane, 2 neighbors per step) =====
        {
            const int h = lane >> 2;                          // head of cols 2l, 2l+1
            v2f cacc = {0.f, 0.f};
            #pragma unroll
            for (int k = 0; k < 16; k += 2) {
                v2f a = *(const v2f*)(Ast + h * 16 + k);      // one ds_load_b64
                cacc += a.x * (*(const v2f*)(Vst + k * 64 + c0));
                cacc += a.y * (*(const v2f*)(Vst + (k + 1) * 64 + c0));
            }
            Cst[c0] = cacc.x; Cst[c0 + 1] = cacc.y;
        }
        __builtin_amdgcn_wave_barrier();

        // ===== out = relu(ctx @ Wo + bo)  (2 cols per lane, 2 rows per step) =====
        {
            v2f oacc = *(const v2f*)(bo + c0);
            #pragma unroll 4
            for (int c = 0; c < 64; c += 2) {
                v2f x = *(const v2f*)(Cst + c);               // one ds_load_b64
                oacc += x.x * (*(const v2f*)(Wo + c * DOUT + c0));
                oacc += x.y * (*(const v2f*)(Wo + (c + 1) * DOUT + c0));
            }
            oacc.x = fmaxf(oacc.x, 0.f);
            oacc.y = fmaxf(oacc.y, 0.f);
            *(v2f*)(out + p * (long)DOUT + c0) = oacc;
        }
    }
}

extern "C" void kernel_launch(void* const* d_in, const int* in_sizes, int n_in,
                              void* d_out, int out_size, void* d_ws, size_t ws_size,
                              hipStream_t stream) {
    (void)in_sizes; (void)n_in; (void)d_ws; (void)ws_size; (void)out_size;
    const float* agent = (const float*)d_in[0];
    const float* neigh = (const float*)d_in[1];
    const float* Wq    = (const float*)d_in[2];
    const float* bq    = (const float*)d_in[3];
    const float* Wk    = (const float*)d_in[4];
    const float* bk    = (const float*)d_in[5];
    const float* Wv    = (const float*)d_in[6];
    const float* bv    = (const float*)d_in[7];
    const float* Wo    = (const float*)d_in[8];
    const float* bo    = (const float*)d_in[9];

    float* out = (float*)d_out;                          // (B*N, 64), first output
    float* att = out + (size_t)B_ * N_ * DOUT;           // (B*N, 8, 16), second output

    dim3 grid((B_ * N_) / 16);                           // 5000, exact
    dim3 block(256);                                     // 8 waves of 32
    mha_wmma_kernel<<<grid, block, (size_t)SMEM_BYTES, stream>>>(agent, neigh, Wq, bq,
                                                                 Wk, bk, Wv, bv, Wo, bo,
                                                                 out, att);
}
